// RNadeMoG_51170240364859
// MI455X (gfx1250) — matrix operand order repaired and checked by reading, not compiled, not run on hardware
//
#include <hip/hip_runtime.h>
#include <stdint.h>

// ---------------- problem constants ----------------
#define BB   512      // batch
#define LL   500      // sequence length (scan dim)
#define HH   512      // hidden
#define DIN  35       // OUT(3)+COND(32)
#define ODIM 35       // 15+15+5 logits per t
#define NC   20       // L-chunks
#define TC   25       // steps per chunk (NC*TC == LL)
#define BT   32       // number of 16-row batch tiles
#define N2P  48       // ODIM padded to 3 WMMA n-tiles
#define HSTR 520      // LDS stride (bf16 elems) for h buffer: dword stride 260 % 64 = 4

typedef __attribute__((ext_vector_type(16))) __bf16 v16bf;
typedef __attribute__((ext_vector_type(8)))  float  v8f;

// workspace layout (u32 units); each fragment = 32 lanes x 8 u32 = 1 KB
static constexpr long WENC_FRAGS = 500L * 64;              // [t][kt(2)][nt(32)]
static constexpr long W2_FRAGS   = 500L * 48;              // [t][kt(16)][nt(3)]
static constexpr long XZ_FRAGS   = 500L * BT * 2;          // [t][btile(32)][kt(2)]
static constexpr long W2B_OFF    = WENC_FRAGS * 256;
static constexpr long XZA_OFF    = W2B_OFF + W2_FRAGS * 256;
static constexpr long S_OFF      = XZA_OFF + XZ_FRAGS * 256;
// total ws = (32000+24000+32000)*1KB + 20*512*512*4B  ~= 111 MB

static __device__ __forceinline__ uint32_t pack_bf2(float lo, float hi) {
  uint32_t a = __float_as_uint(lo), b = __float_as_uint(hi);
  uint32_t ra = (a + 0x7FFFu + ((a >> 16) & 1u)) >> 16;   // RNE f32->bf16
  uint32_t rb = (b + 0x7FFFu + ((b >> 16) & 1u)) >> 16;
  return (ra & 0xFFFFu) | (rb << 16);
}
static __device__ __forceinline__ unsigned short f2bf(float f) {
  uint32_t a = __float_as_uint(f);
  return (unsigned short)((a + 0x7FFFu + ((a >> 16) & 1u)) >> 16);
}

union FragB { uint4 q[2]; v16bf v; };
union FragA { uint32_t u[8]; v16bf v; };

// fragment stored lane-major (32 lanes x 8 u32); works for both A and B fragments
static __device__ __forceinline__ v16bf load_frag(const uint32_t* frag, int lane) {
  FragB f;
  const uint4* p = (const uint4*)frag;
  f.q[0] = p[lane * 2 + 0];
  f.q[1] = p[lane * 2 + 1];
  return f.v;
}

// A fragment (16x32 bf16) gathered from bf16 LDS h-buffer; K-pairs are contiguous u32
static __device__ __forceinline__ v16bf gather_h(const unsigned short* hb, int kt32, int lane) {
  const int m  = lane & 15;
  const int kh = (lane >> 4) << 3;
  FragA f;
#pragma unroll
  for (int i = 0; i < 8; ++i) {
    const int K = kt32 + (kh << 1) + (i << 1);
    f.u[i] = *(const uint32_t*)(hb + m * HSTR + K);
  }
  return f.v;
}

static __device__ __forceinline__ v8f wmma_bf16(v16bf a, v16bf b, v8f c) {
  return __builtin_amdgcn_wmma_f32_16x16x32_bf16(false, a, false, b, (short)0, c,
                                                 false, false);
}

static __device__ __forceinline__ float w2col(const float* wm, const float* wsg,
                                              const float* wp, int t, int K, int N) {
  if (N < 15) return wm[((long)t * HH + K) * 15 + N];
  if (N < 30) return wsg[((long)t * HH + K) * 15 + (N - 15)];
  if (N < 35) return wp[((long)t * HH + K) * 5 + (N - 30)];
  return 0.f;
}
static __device__ __forceinline__ float xzval(const float* x, const float* z,
                                              int b, int t, int col) {
  if (col < 3)  return x[((long)b * LL + t) * 3 + col];
  if (col < 35) return z[((long)b * LL + t) * 32 + (col - 3)];
  return 0.f;
}

// ---------------- kernel P: pack everything static into bf16 WMMA fragments --------
// B frag (32x16): lane<16 -> col N=lane, K 0..15 pairs; lane>=16 -> col N=lane-16, K 16..31
// A frag (16x32): lane<16 -> row M=lane, K 0..15 pairs; lane>=16 -> row M=lane-16, K 16..31
__global__ void __launch_bounds__(256)
pack_kernel(const float* __restrict__ Wenc, const float* __restrict__ wm,
            const float* __restrict__ wsg, const float* __restrict__ wp,
            const float* __restrict__ x, const float* __restrict__ z,
            uint32_t* __restrict__ ws) {
  const int lane = threadIdx.x & 31;
  const int wv   = threadIdx.x >> 5;
  const long f   = (long)blockIdx.x * 8 + wv;   // one fragment per wave
  uint32_t u[8];
  uint4* dst;
  if (f < WENC_FRAGS) {
    const int t  = (int)(f >> 6);
    const int kt = (int)((f >> 5) & 1);
    const int nt = (int)(f & 31);
    const int N  = nt * 16 + (lane & 15);
    const int kb = kt * 32 + ((lane >> 4) << 4);
#pragma unroll
    for (int i = 0; i < 8; ++i) {
      const int K = kb + 2 * i;
      const float lo = (K     < DIN) ? Wenc[((long)t * DIN + K)     * HH + N] : 0.f;
      const float hi = (K + 1 < DIN) ? Wenc[((long)t * DIN + K + 1) * HH + N] : 0.f;
      u[i] = pack_bf2(lo, hi);
    }
    dst = (uint4*)(ws + f * 256 + lane * 8);
  } else if (f < WENC_FRAGS + W2_FRAGS) {
    const long g = f - WENC_FRAGS;
    const int t  = (int)(g / 48);
    const int r  = (int)(g % 48);
    const int kt = r / 3, nt = r % 3;
    const int N  = nt * 16 + (lane & 15);
    const int kb = kt * 32 + ((lane >> 4) << 4);
#pragma unroll
    for (int i = 0; i < 8; ++i) {
      const int K = kb + 2 * i;
      u[i] = pack_bf2(w2col(wm, wsg, wp, t, K, N), w2col(wm, wsg, wp, t, K + 1, N));
    }
    dst = (uint4*)(ws + W2B_OFF + g * 256 + lane * 8);
  } else {
    const long g = f - WENC_FRAGS - W2_FRAGS;   // xz A-fragment [t][btile][kt]
    if (g >= XZ_FRAGS) return;
    const int t     = (int)(g >> 6);
    const int btile = (int)((g >> 1) & 31);
    const int kt    = (int)(g & 1);
    const int b     = btile * 16 + (lane & 15);
    const int kh    = (lane >> 4) << 3;
#pragma unroll
    for (int i = 0; i < 8; ++i) {
      const int K = kt * 32 + (kh << 1) + (i << 1);
      u[i] = pack_bf2(xzval(x, z, b, t, K), xzval(x, z, b, t, K + 1));
    }
    dst = (uint4*)(ws + XZA_OFF + g * 256 + lane * 8);
  }
  dst[0] = make_uint4(u[0], u[1], u[2], u[3]);
  dst[1] = make_uint4(u[4], u[5], u[6], u[7]);
}

// ---------------- phase 1: per-chunk sums  S[c] = sum_{t in chunk} xz_t @ Wenc_t ----
// pure global_load_b128 -> v_wmma stream: no LDS, no barriers, no conversions.
// All 10 fragments of a step are loaded into distinct live values first so the
// compiler can clause the loads and use partial s_wait_loadcnt under the WMMAs.
__global__ void __launch_bounds__(256)
phase1_chunk_sums(const uint32_t* __restrict__ wsu, float* __restrict__ S) {
  const int btile = blockIdx.x, chunk = blockIdx.y;
  const int lane = threadIdx.x & 31, wv = threadIdx.x >> 5;
  const uint32_t* xza = wsu + XZA_OFF;
  const v8f zero8 = {0.f, 0.f, 0.f, 0.f, 0.f, 0.f, 0.f, 0.f};
  v8f acc[4] = {zero8, zero8, zero8, zero8};
  const int t0 = chunk * TC, t1 = t0 + TC;
  for (int t = t0; t < t1; ++t) {
    const long fb = ((long)t * BT + btile) * 2;
    const v16bf a0 = load_frag(xza + (fb + 0) * 256, lane);
    const v16bf a1 = load_frag(xza + (fb + 1) * 256, lane);
    v16bf bb[8];
#pragma unroll
    for (int j = 0; j < 4; ++j) {
      const int nt = wv * 4 + j;
      bb[j]     = load_frag(wsu + (((long)t * 2 + 0) * 32 + nt) * 256, lane);
      bb[4 + j] = load_frag(wsu + (((long)t * 2 + 1) * 32 + nt) * 256, lane);
    }
#pragma unroll
    for (int j = 0; j < 4; ++j) acc[j] = wmma_bf16(a0, bb[j], acc[j]);
#pragma unroll
    for (int j = 0; j < 4; ++j) acc[j] = wmma_bf16(a1, bb[4 + j], acc[j]);
    if (t + 1 < t1) {
      __builtin_prefetch(wsu + (((long)(t + 1) * 2) * 32 + wv * 4) * 256, 0, 3);
      __builtin_prefetch(xza + ((long)(t + 1) * BT + btile) * 2 * 256, 0, 3);
    }
  }
  const int kh8 = (lane >> 4) << 3, n0 = lane & 15;
#pragma unroll
  for (int j = 0; j < 4; ++j) {
    const int nt = wv * 4 + j;
#pragma unroll
    for (int r = 0; r < 8; ++r)
      S[((long)chunk * BB + btile * 16 + r + kh8) * HH + nt * 16 + n0] = acc[j][r];
  }
}

// ---------------- phase 2: exclusive prefix over chunks (+ b_enc/r0) ----------------
__global__ void __launch_bounds__(256)
phase2_prefix(const float* __restrict__ benc, const float* __restrict__ resc,
              float* __restrict__ S) {
  const int idx = blockIdx.x * 256 + threadIdx.x;   // over B*H
  float val = benc[idx & (HH - 1)] / resc[0];
  for (int c = 0; c < NC; ++c) {
    float* p = S + (long)c * BB * HH + idx;
    const float tmp = *p;
    *p = val;            // carry-in c_t at chunk start
    val += tmp;
  }
}

// ---------------- phase 3: replay chunk with carry, emit logits ----------------
__global__ void __launch_bounds__(256)
phase3_main(const float* __restrict__ bm, const float* __restrict__ bs,
            const float* __restrict__ bp, const float* __restrict__ resc,
            const uint32_t* __restrict__ wsu, const float* __restrict__ S,
            float* __restrict__ out) {
  __shared__ unsigned short hbuf[16 * HSTR];        // bf16 h snapshot
  __shared__ float parts[8 * 16 * N2P];
  const int btile = blockIdx.x, chunk = blockIdx.y;
  const int lane = threadIdx.x & 31, wv = threadIdx.x >> 5;
  const int kh8 = (lane >> 4) << 3, n0 = lane & 15;
  const uint32_t* W2B = wsu + W2B_OFF;
  const uint32_t* xza = wsu + XZA_OFF;
  const v8f zero8 = {0.f, 0.f, 0.f, 0.f, 0.f, 0.f, 0.f, 0.f};
  v8f acc[4];
#pragma unroll
  for (int j = 0; j < 4; ++j) {       // load carry into accumulator layout
    const int nt = wv * 4 + j;
#pragma unroll
    for (int r = 0; r < 8; ++r)
      acc[j][r] = S[((long)chunk * BB + btile * 16 + r + kh8) * HH + nt * 16 + n0];
  }
  const int t0 = chunk * TC, t1 = t0 + TC;
  for (int t = t0; t < t1; ++t) {
    __syncthreads();                                  // A: prev gathers/reduce done
    const float rt = resc[t];
#pragma unroll
    for (int j = 0; j < 4; ++j) {                     // snapshot h = relu(r_t*c) as bf16
      const int nt = wv * 4 + j;
#pragma unroll
      for (int r = 0; r < 8; ++r) {
        float hv = rt * acc[j][r];
        hv = hv > 0.f ? hv : 0.f;
        hbuf[(r + kh8) * HSTR + nt * 16 + n0] = f2bf(hv);
      }
    }
    __syncthreads();                                  // B: h visible
    // GEMM2: K split across waves (2 ktiles each); batch all loads before WMMAs
    const v16bf ah0 = gather_h(hbuf, (wv * 2 + 0) * 32, lane);
    const v16bf ah1 = gather_h(hbuf, (wv * 2 + 1) * 32, lane);
    v16bf wb[6];
#pragma unroll
    for (int kk = 0; kk < 2; ++kk)
#pragma unroll
      for (int j = 0; j < 3; ++j)
        wb[kk * 3 + j] =
            load_frag(W2B + (((long)t * 16 + wv * 2 + kk) * 3 + j) * 256, lane);
    v8f p2[3] = {zero8, zero8, zero8};
#pragma unroll
    for (int j = 0; j < 3; ++j) p2[j] = wmma_bf16(ah0, wb[j], p2[j]);
#pragma unroll
    for (int j = 0; j < 3; ++j) p2[j] = wmma_bf16(ah1, wb[3 + j], p2[j]);
#pragma unroll
    for (int j = 0; j < 3; ++j)
#pragma unroll
      for (int r = 0; r < 8; ++r)
        parts[(wv * 16 + r + kh8) * N2P + j * 16 + n0] = p2[j][r];
    // GEMM1: c += xz_t @ Wenc_t (batched loads, state stays in accumulators)
    {
      const long fb = ((long)t * BT + btile) * 2;
      const v16bf a0 = load_frag(xza + (fb + 0) * 256, lane);
      const v16bf a1 = load_frag(xza + (fb + 1) * 256, lane);
      v16bf bb[8];
#pragma unroll
      for (int j = 0; j < 4; ++j) {
        const int nt = wv * 4 + j;
        bb[j]     = load_frag(wsu + (((long)t * 2 + 0) * 32 + nt) * 256, lane);
        bb[4 + j] = load_frag(wsu + (((long)t * 2 + 1) * 32 + nt) * 256, lane);
      }
#pragma unroll
      for (int j = 0; j < 4; ++j) acc[j] = wmma_bf16(a0, bb[j], acc[j]);
#pragma unroll
      for (int j = 0; j < 4; ++j) acc[j] = wmma_bf16(a1, bb[4 + j], acc[j]);
    }
    if (t + 1 < t1) {
      __builtin_prefetch(wsu + (((long)(t + 1) * 2) * 32 + wv * 4) * 256, 0, 3);
      __builtin_prefetch(W2B + (((long)(t + 1) * 16 + wv * 2) * 3) * 256, 0, 3);
    }
    __syncthreads();                                  // C: partials visible
    // deterministic serial reduction over 8 waves + bias, write logits
    for (int i = threadIdx.x; i < 16 * N2P; i += 256) {
      const int M = i / N2P, N = i % N2P;
      if (N < ODIM) {
        float s;
        if (N < 15)      s = bm[(long)t * 15 + N];
        else if (N < 30) s = bs[(long)t * 15 + (N - 15)];
        else             s = bp[(long)t * 5 + (N - 30)];
#pragma unroll
        for (int w = 0; w < 8; ++w) s += parts[(w * 16 + M) * N2P + N];
        out[(((long)btile * 16 + M) * LL + t) * ODIM + N] = s;
      }
    }
  }
}

// ---------------- host entry ----------------
extern "C" void kernel_launch(void* const* d_in, const int* in_sizes, int n_in,
                              void* d_out, int out_size, void* d_ws, size_t ws_size,
                              hipStream_t stream) {
  (void)in_sizes; (void)n_in; (void)out_size; (void)ws_size;
  const float* x    = (const float*)d_in[0];
  const float* z    = (const float*)d_in[1];
  const float* Wenc = (const float*)d_in[2];
  const float* benc = (const float*)d_in[3];
  const float* wm   = (const float*)d_in[4];
  const float* bm   = (const float*)d_in[5];
  const float* wsg  = (const float*)d_in[6];
  const float* bs   = (const float*)d_in[7];
  const float* wp   = (const float*)d_in[8];
  const float* bp   = (const float*)d_in[9];
  const float* resc = (const float*)d_in[10];
  uint32_t* wsu = (uint32_t*)d_ws;
  float* S      = (float*)(wsu + S_OFF);
  float* out    = (float*)d_out;

  const long totalFrags = WENC_FRAGS + W2_FRAGS + XZ_FRAGS;       // 88000
  pack_kernel<<<(int)(totalFrags / 8), 256, 0, stream>>>(Wenc, wm, wsg, wp, x, z, wsu);
  phase1_chunk_sums<<<dim3(BT, NC), 256, 0, stream>>>(wsu, S);
  phase2_prefix<<<(BB * HH) / 256, 256, 0, stream>>>(benc, resc, S);
  phase3_main<<<dim3(BT, NC), 256, 0, stream>>>(bm, bs, bp, resc, wsu, S, out);
}
